// MultiHeadSelfAttention_34428457844780
// MI455X (gfx1250) — matrix operand order repaired and checked by reading, not compiled, not run on hardware
//
#include <hip/hip_runtime.h>

// ---------------------------------------------------------------------------
// Multi-head self-attention forward for MI455X (gfx1250, wave32, WMMA).
//   B=4, S=2048, E=1024, H=16, D=64.
// Pipeline:
//   1) convert query/w_in/w_out fp32 -> f16 staging (L2-resident)
//   2) QKV GEMM (f16 WMMA, f32 acc) + bias, scatter to Q[B,H,S,D] (pre-scaled
//      by 1/sqrt(D)), K[B,H,S,D], V^T[B,H,D,S]
//   3) causal flash-attention: async global->LDS double-buffered K/V tiles,
//      per-wave 16-row WMMA scores + WMMA P@V
//   4) output projection GEMM + bias -> fp32 d_out
// ---------------------------------------------------------------------------

typedef __attribute__((ext_vector_type(16))) _Float16 v16h;
typedef __attribute__((ext_vector_type(8)))  _Float16 v8h;
typedef __attribute__((ext_vector_type(4)))  _Float16 v4h;
typedef __attribute__((ext_vector_type(8)))  float    v8f;

#define B_  4
#define S_  2048
#define E_  1024
#define H_  16
#define D_  64
#define F_  3072            // 3*E
#define M_  8192            // B*S
#define LOG2E 1.44269504f
#define KV_LD 72            // padded LDS row stride (halves): 144B = 36 banks

static __device__ __forceinline__ v16h cat8(v8h lo, v8h hi) {
  return __builtin_shufflevector(lo, hi, 0,1,2,3,4,5,6,7,8,9,10,11,12,13,14,15);
}

static __device__ __forceinline__ v8f wmma_f16(v16h a, v16h b, v8f c) {
  // v_wmma_f32_16x16x32_f16  D = A*B + C   (fp32 accumulate)
  return __builtin_amdgcn_wmma_f32_16x16x32_f16(false, a, false, b,
                                                (short)0, c, false, false);
}

// async copy of one 16-byte chunk: global -> LDS, tracked by ASYNCcnt.
// lds_off = low 32 bits of generic shared pointer (== LDS byte offset),
// gaddr   = 64-bit global address in a VGPR pair, saddr = off.
static __device__ __forceinline__ void async_g2l_b128(const void* lds_p, const void* g_p) {
  unsigned int       lds_off = (unsigned int)(unsigned long long)(size_t)lds_p;
  unsigned long long gaddr   = (unsigned long long)(size_t)g_p;
  asm volatile("global_load_async_to_lds_b128 %0, %1, off"
               :: "v"(lds_off), "v"(gaddr) : "memory");
}

static __device__ __forceinline__ void wait_asynccnt0() {
  asm volatile("s_wait_asynccnt 0" ::: "memory");
}

// A-fragment (16x32, MxK), row-major source with leading dim `ld` (elements).
// Layout used here (contiguous k): lane L (L<16): M=L, K = k0+[0..15];
//             lane L (>=16): M=L-16, K = k0+[16..31].
static __device__ __forceinline__ v16h load_A_frag(const _Float16* __restrict__ base,
                                                   int ld, int row0, int k0, int lane) {
  int r  = row0 + (lane & 15);
  int kk = k0 + ((lane >> 4) << 4);
  const _Float16* p = base + (size_t)r * ld + kk;
  v8h lo = *(const v8h*)p;
  v8h hi = *(const v8h*)(p + 8);
  return cat8(lo, hi);
}

// B-fragment (32x16, KxN) for C = A * W^T where W rows run along K:
// B[k][n] = W[n0+n][k].  Layout used here (contiguous k): lane L: N = L&15, K = (L>>4)*16 + j,
// so each lane reads 16 contiguous halves of one W row.
static __device__ __forceinline__ v16h load_B_fragT(const _Float16* __restrict__ base,
                                                    int ld, int n0, int k0, int lane) {
  int n  = n0 + (lane & 15);
  int kk = k0 + ((lane >> 4) << 4);
  const _Float16* p = base + (size_t)n * ld + kk;
  v8h lo = *(const v8h*)p;
  v8h hi = *(const v8h*)(p + 8);
  return cat8(lo, hi);
}

// B-fragment straight out of a padded LDS tile (rows along N, K contiguous).
static __device__ __forceinline__ v16h load_B_frag_lds(const _Float16* tile,
                                                       int n_in_tile, int k0, int lane) {
  const _Float16* p = tile + (n_in_tile + (lane & 15)) * KV_LD + k0 + ((lane >> 4) << 4);
  v8h lo = *(const v8h*)p;
  v8h hi = *(const v8h*)(p + 8);
  return cat8(lo, hi);
}

// ---------------------------------------------------------------------------
__global__ void cvt_f32_to_f16(const float* __restrict__ s,
                               _Float16* __restrict__ d, int n) {
  int i = (blockIdx.x * blockDim.x + threadIdx.x) * 4;
  if (i + 3 < n) {
    float4 v = *(const float4*)(s + i);
    v4h h = { (_Float16)v.x, (_Float16)v.y, (_Float16)v.z, (_Float16)v.w };
    *(v4h*)(d + i) = h;
  }
}

// ---------------------------------------------------------------------------
// QKV projection: C[m,f] = sum_e X[m,e]*Win[f,e] + b_in[f]; scatter epilogue.
// Block tile 128x128, 8 waves in a 2(M)x4(N) grid; wave tile 64x32.
__global__ __launch_bounds__(256)
void qkv_gemm(const _Float16* __restrict__ X,    // [M_,E_]
              const _Float16* __restrict__ W,    // [F_,E_]
              const float*    __restrict__ bias, // [F_]
              _Float16* __restrict__ Qh,         // [B,H,S,D], pre-scaled
              _Float16* __restrict__ Kh,         // [B,H,S,D]
              _Float16* __restrict__ Vt)         // [B,H,D,S]
{
  int wid  = threadIdx.x >> 5;
  int lane = threadIdx.x & 31;
  int m0 = blockIdx.x * 128 + (wid >> 2) * 64;
  int f0 = blockIdx.y * 128 + (wid & 3) * 32;

  v8f acc[4][2] = {};
  for (int k0 = 0; k0 < E_; k0 += 32) {
    v16h b0 = load_B_fragT(W, E_, f0,      k0, lane);
    v16h b1 = load_B_fragT(W, E_, f0 + 16, k0, lane);
#pragma unroll
    for (int mf = 0; mf < 4; ++mf) {
      v16h a = load_A_frag(X, E_, m0 + mf * 16, k0, lane);
      acc[mf][0] = wmma_f16(a, b0, acc[mf][0]);
      acc[mf][1] = wmma_f16(a, b1, acc[mf][1]);
    }
  }

#pragma unroll
  for (int mf = 0; mf < 4; ++mf)
#pragma unroll
    for (int nf = 0; nf < 2; ++nf)
#pragma unroll
      for (int r = 0; r < 8; ++r) {
        int m = m0 + mf * 16 + r + ((lane >> 4) << 3);
        int f = f0 + nf * 16 + (lane & 15);
        float v = acc[mf][nf][r] + bias[f];
        int b  = m >> 11;          // m / S_
        int s  = m & (S_ - 1);
        int t  = f >> 10;          // q/k/v third
        int fe = f & (E_ - 1);
        int h  = fe >> 6;
        int d  = fe & (D_ - 1);
        if (t == 0)      Qh[(((size_t)(b * H_ + h)) * S_ + s) * D_ + d] = (_Float16)(v * 0.125f);
        else if (t == 1) Kh[(((size_t)(b * H_ + h)) * S_ + s) * D_ + d] = (_Float16)v;
        else             Vt[(((size_t)(b * H_ + h)) * D_ + d) * S_ + s] = (_Float16)v;
      }
}

// ---------------------------------------------------------------------------
// Causal flash-attention. Grid: (S/128, B*H); 8 waves x 16 query rows each.
// K/V 64-key tiles are staged cooperatively into LDS with async global->LDS
// copies (ASYNCcnt), double-buffered so the DMA of block jc+1 overlaps the
// WMMA work on block jc.  All waves run a uniform kv-block loop (barriers are
// EXEC-uniform); waves past their causal limit skip only the compute.
__global__ __launch_bounds__(256)
void attn_fwd(const _Float16* __restrict__ Qh,  // [B,H,S,D]
              const _Float16* __restrict__ Kh,  // [B,H,S,D]
              const _Float16* __restrict__ Vt,  // [B,H,D,S]
              _Float16* __restrict__ Ctx)       // [B,S,E]
{
  __shared__ __align__(16) _Float16 Ktile[2][64 * KV_LD];  // [key][d]
  __shared__ __align__(16) _Float16 Vtile[2][64 * KV_LD];  // [d][key]
  __shared__ __align__(16) _Float16 Ps[8][16 * KV_LD];     // per-wave P tile

  int tid  = threadIdx.x;
  int wid  = tid >> 5;
  int lane = tid & 31;
  int bh = blockIdx.y;
  int b = bh >> 4, h = bh & 15;
  int q0 = blockIdx.x * 128 + wid * 16;          // wave's first query row

  const _Float16* Qbh = Qh + (size_t)bh * S_ * D_;
  const _Float16* Kbh = Kh + (size_t)bh * S_ * D_;
  const _Float16* Vbh = Vt + (size_t)bh * D_ * S_;

  v16h qf[2];
  qf[0] = load_A_frag(Qbh, D_, q0, 0,  lane);
  qf[1] = load_A_frag(Qbh, D_, q0, 32, lane);

  v8f o[4] = {};
  float mrow[8], lrow[8];
#pragma unroll
  for (int r = 0; r < 8; ++r) { mrow[r] = -3.0e38f; lrow[r] = 0.0f; }

  _Float16* myP = &Ps[wid][0];
  int jcend = (blockIdx.x * 128 + 127) >> 6;     // uniform across block
  int myjc  = (q0 + 15) >> 6;                    // this wave's causal limit

  // issue async K/V tile DMA for kv-block jc into buffer bu:
  // 512 16B chunks per tile, 2 per thread per tile.
  auto issue_tiles = [&](int jc, int bu) {
    int c0 = jc << 6;
#pragma unroll
    for (int i = 0; i < 2; ++i) {
      int chunk = tid + i * 256;            // 0..511
      int row   = chunk >> 3;               // 0..63
      int col   = (chunk & 7) << 3;         // halves
      async_g2l_b128(&Ktile[bu][row * KV_LD + col],
                     Kbh + (size_t)(c0 + row) * D_ + col);
      async_g2l_b128(&Vtile[bu][row * KV_LD + col],
                     Vbh + (size_t)row * S_ + c0 + col);
    }
  };

  issue_tiles(0, 0);
  for (int jc = 0; jc <= jcend; ++jc) {
    int bu = jc & 1;
    int c0 = jc << 6;
    wait_asynccnt0();        // my DMA for buffer bu complete
    __syncthreads();         // everyone's DMA done; previous readers of bu^1 done
    if (jc < jcend) issue_tiles(jc + 1, bu ^ 1);

    if (jc <= myjc) {
      // ---- scores S = Q K^T from LDS K tile ----
      v8f sc[4];
#pragma unroll
      for (int nf = 0; nf < 4; ++nf) {
        v16h k0f = load_B_frag_lds(&Ktile[bu][0], nf * 16, 0,  lane);
        v16h k1f = load_B_frag_lds(&Ktile[bu][0], nf * 16, 32, lane);
        v8f t = {};
        t = wmma_f16(qf[0], k0f, t);
        t = wmma_f16(qf[1], k1f, t);
        sc[nf] = t;
      }

      // ---- causal mask (diagonal blocks only) ----
      if (c0 + 63 > q0) {
#pragma unroll
        for (int nf = 0; nf < 4; ++nf) {
          int col = c0 + nf * 16 + (lane & 15);
#pragma unroll
          for (int r = 0; r < 8; ++r) {
            int m = q0 + r + ((lane >> 4) << 3);
            if (col > m) sc[nf][r] = -3.0e38f;
          }
        }
      }

      // ---- online softmax (rows live in 16-lane half-groups) ----
#pragma unroll
      for (int r = 0; r < 8; ++r) {
        float mx = fmaxf(fmaxf(sc[0][r], sc[1][r]), fmaxf(sc[2][r], sc[3][r]));
#pragma unroll
        for (int w = 1; w < 16; w <<= 1) mx = fmaxf(mx, __shfl_xor(mx, w, 32));
        float mnew  = fmaxf(mrow[r], mx);
        float alpha = exp2f((mrow[r] - mnew) * LOG2E);
        float rs = 0.0f;
#pragma unroll
        for (int nf = 0; nf < 4; ++nf) {
          float p = exp2f((sc[nf][r] - mnew) * LOG2E);
          sc[nf][r] = p;
          rs += p;
        }
#pragma unroll
        for (int w = 1; w < 16; w <<= 1) rs += __shfl_xor(rs, w, 32);
        lrow[r] = lrow[r] * alpha + rs;
        mrow[r] = mnew;
#pragma unroll
        for (int nf2 = 0; nf2 < 4; ++nf2) o[nf2][r] *= alpha;
      }

      // ---- C-layout -> A-layout transpose of P via private LDS tile ----
#pragma unroll
      for (int nf = 0; nf < 4; ++nf) {
        int col = nf * 16 + (lane & 15);
#pragma unroll
        for (int r = 0; r < 8; ++r) {
          int m = r + ((lane >> 4) << 3);
          myP[m * KV_LD + col] = (_Float16)sc[nf][r];
        }
      }
      v16h pf[2];
#pragma unroll
      for (int ks = 0; ks < 2; ++ks) {
        const _Float16* pp = myP + (lane & 15) * KV_LD + ks * 32 + ((lane >> 4) << 4);
        v8h lo = *(const v8h*)pp;
        v8h hi = *(const v8h*)(pp + 8);
        pf[ks] = cat8(lo, hi);
      }

      // ---- O += P @ V from LDS V^T tile ----
#pragma unroll
      for (int nf2 = 0; nf2 < 4; ++nf2)
#pragma unroll
        for (int ks = 0; ks < 2; ++ks)
          o[nf2] = wmma_f16(pf[ks],
                            load_B_frag_lds(&Vtile[bu][0], nf2 * 16, ks * 32, lane),
                            o[nf2]);
    }
    __syncthreads();         // all reads of buffer bu done before it is reused
  }

  // ---- normalize and write context in [B,S,E] f16 ----
#pragma unroll
  for (int nf2 = 0; nf2 < 4; ++nf2)
#pragma unroll
    for (int r = 0; r < 8; ++r) {
      int s  = q0 + r + ((lane >> 4) << 3);
      int dd = h * D_ + nf2 * 16 + (lane & 15);
      Ctx[((size_t)(b * S_ + s)) * E_ + dd] = (_Float16)(o[nf2][r] / lrow[r]);
    }
}

// ---------------------------------------------------------------------------
// Output projection: out[m,n] = sum_e Ctx[m,e]*Wout[n,e] + b_out[n]  (fp32 out)
__global__ __launch_bounds__(256)
void out_gemm(const _Float16* __restrict__ Ctx,  // [M_,E_]
              const _Float16* __restrict__ W,    // [E_,E_]
              const float*    __restrict__ bias, // [E_]
              float* __restrict__ out)           // [M_,E_]
{
  int wid  = threadIdx.x >> 5;
  int lane = threadIdx.x & 31;
  int m0 = blockIdx.x * 128 + (wid >> 2) * 64;
  int f0 = blockIdx.y * 128 + (wid & 3) * 32;

  v8f acc[4][2] = {};
  for (int k0 = 0; k0 < E_; k0 += 32) {
    v16h b0 = load_B_fragT(W, E_, f0,      k0, lane);
    v16h b1 = load_B_fragT(W, E_, f0 + 16, k0, lane);
#pragma unroll
    for (int mf = 0; mf < 4; ++mf) {
      v16h a = load_A_frag(Ctx, E_, m0 + mf * 16, k0, lane);
      acc[mf][0] = wmma_f16(a, b0, acc[mf][0]);
      acc[mf][1] = wmma_f16(a, b1, acc[mf][1]);
    }
  }
#pragma unroll
  for (int mf = 0; mf < 4; ++mf)
#pragma unroll
    for (int nf = 0; nf < 2; ++nf)
#pragma unroll
      for (int r = 0; r < 8; ++r) {
        int m = m0 + mf * 16 + r + ((lane >> 4) << 3);
        int f = f0 + nf * 16 + (lane & 15);
        out[(size_t)m * E_ + f] = acc[mf][nf][r] + bias[f];
      }
}

// ---------------------------------------------------------------------------
extern "C" void kernel_launch(void* const* d_in, const int* in_sizes, int n_in,
                              void* d_out, int out_size, void* d_ws, size_t ws_size,
                              hipStream_t stream) {
  const float* q_f32  = (const float*)d_in[0];   // [B,S,E]
  const float* wi_f32 = (const float*)d_in[1];   // [3E,E]
  const float* bi     = (const float*)d_in[2];   // [3E]
  const float* wo_f32 = (const float*)d_in[3];   // [E,E]
  const float* bo     = (const float*)d_in[4];   // [E]
  float* out = (float*)d_out;

  char* ws = (char*)d_ws;
  _Float16* Xh  = (_Float16*)ws; ws += (size_t)M_ * E_ * 2;
  _Float16* Wih = (_Float16*)ws; ws += (size_t)F_ * E_ * 2;
  _Float16* Woh = (_Float16*)ws; ws += (size_t)E_ * E_ * 2;
  _Float16* Qh  = (_Float16*)ws; ws += (size_t)M_ * E_ * 2;
  _Float16* Kh  = (_Float16*)ws; ws += (size_t)M_ * E_ * 2;
  _Float16* Vt  = (_Float16*)ws; ws += (size_t)M_ * E_ * 2;
  _Float16* Ctx = (_Float16*)ws; ws += (size_t)M_ * E_ * 2;

  cvt_f32_to_f16<<<(M_ * E_) / 1024, 256, 0, stream>>>(q_f32,  Xh,  M_ * E_);
  cvt_f32_to_f16<<<(F_ * E_) / 1024, 256, 0, stream>>>(wi_f32, Wih, F_ * E_);
  cvt_f32_to_f16<<<(E_ * E_) / 1024, 256, 0, stream>>>(wo_f32, Woh, E_ * E_);

  qkv_gemm<<<dim3(M_ / 128, F_ / 128), 256, 0, stream>>>(Xh, Wih, bi, Qh, Kh, Vt);
  attn_fwd<<<dim3(S_ / 128, B_ * H_), 256, 0, stream>>>(Qh, Kh, Vt, Ctx);
  out_gemm<<<dim3(M_ / 128, E_ / 128), 256, 0, stream>>>(Ctx, Woh, bo, out);
}
